// MambaAttentionDecoder_74466142978810
// MI455X (gfx1250) — matrix operand order repaired and checked by reading, not compiled, not run on hardware
//
#include <hip/hip_runtime.h>

#define B_   64
#define L_   256
#define T_   32
#define E_   384
#define H_   6
#define HD_  64
#define DI_  768
#define DS_  16
#define DC_  4
#define DR_  24
#define HID_ 1536
#define V_   97
#define ND_  12
#define EPS_ 1e-5f

typedef __bf16 bf16_t;
typedef __attribute__((ext_vector_type(16))) __bf16 v16bf;
typedef __attribute__((ext_vector_type(8)))  float  v8f;
typedef __attribute__((ext_vector_type(4)))  unsigned int v4u;
typedef __attribute__((ext_vector_type(8)))  int    v8i;
typedef __attribute__((ext_vector_type(4)))  int    v4i;

#if defined(__has_builtin)
#if __has_builtin(__builtin_amdgcn_tensor_load_to_lds) && __has_builtin(__builtin_amdgcn_s_wait_tensorcnt)
#define USE_TDM 1
#endif
#endif
#ifndef USE_TDM
#define USE_TDM 0
#endif

__device__ __forceinline__ float siluf(float x) { return x / (1.f + __expf(-x)); }

// ---------------------------------------------------------------------------
// f32 -> bf16 weight conversion
// ---------------------------------------------------------------------------
__global__ void k_cvt(const float* __restrict__ s, bf16_t* __restrict__ d, int n) {
  int i = blockIdx.x * blockDim.x + threadIdx.x;
  if (i < n) d[i] = (bf16_t)s[i];
}

// hidden = query_embed + pos_emb (broadcast over batch); residual = 0
__global__ void k_init(const float* __restrict__ qe, const float* __restrict__ pe,
                       float* __restrict__ hidden, float* __restrict__ residual) {
  int i = blockIdx.x * blockDim.x + threadIdx.x;
  if (i >= B_ * T_ * E_) return;
  int te = i % (T_ * E_);
  hidden[i]   = qe[te] + pe[te];
  residual[i] = 0.f;
}

__global__ void k_add(float* __restrict__ d, const float* __restrict__ s, int n) {
  int i = blockIdx.x * blockDim.x + threadIdx.x;
  if (i < n) d[i] += s[i];
}

// residual += hidden; out = rmsnorm(residual) * w   (one row of E per block)
__global__ __launch_bounds__(256) void k_add_rmsnorm(float* __restrict__ residual,
    const float* __restrict__ hidden, const float* __restrict__ w,
    float* __restrict__ out) {
  int r = blockIdx.x;
  __shared__ float red[256];
  float ss = 0.f;
  for (int c = threadIdx.x; c < E_; c += 256) {
    size_t i = (size_t)r * E_ + c;
    float v = residual[i] + hidden[i];
    residual[i] = v;
    ss += v * v;
  }
  red[threadIdx.x] = ss;
  __syncthreads();
  for (int st = 128; st > 0; st >>= 1) {
    if (threadIdx.x < st) red[threadIdx.x] += red[threadIdx.x + st];
    __syncthreads();
  }
  float rstd = rsqrtf(red[0] / (float)E_ + EPS_);
  for (int c = threadIdx.x; c < E_; c += 256) {
    size_t i = (size_t)r * E_ + c;
    out[i] = residual[i] * rstd * w[c];
  }
}

__global__ __launch_bounds__(256) void k_layernorm(const float* __restrict__ x,
    const float* __restrict__ w, const float* __restrict__ b,
    float* __restrict__ out) {
  int r = blockIdx.x;
  __shared__ float rs[256], rq[256];
  float s = 0.f, q = 0.f;
  for (int c = threadIdx.x; c < E_; c += 256) {
    float v = x[(size_t)r * E_ + c];
    s += v; q += v * v;
  }
  rs[threadIdx.x] = s; rq[threadIdx.x] = q;
  __syncthreads();
  for (int st = 128; st > 0; st >>= 1) {
    if (threadIdx.x < st) { rs[threadIdx.x] += rs[threadIdx.x + st]; rq[threadIdx.x] += rq[threadIdx.x + st]; }
    __syncthreads();
  }
  float mu = rs[0] / (float)E_;
  float var = rq[0] / (float)E_ - mu * mu;
  float rstd = rsqrtf(var + EPS_);
  for (int c = threadIdx.x; c < E_; c += 256) {
    size_t i = (size_t)r * E_ + c;
    out[i] = (x[i] - mu) * rstd * w[c] + b[c];
  }
}

// depthwise causal conv (DC=4) + bias + silu ; input = xm half of xz
__global__ void k_conv_silu(const float* __restrict__ xz, const float* __restrict__ cw,
                            const float* __restrict__ cb, float* __restrict__ xc) {
  int g = blockIdx.x * blockDim.x + threadIdx.x;
  if (g >= B_ * T_ * DI_) return;
  int d = g % DI_;
  int t = (g / DI_) % T_;
  int b = g / (DI_ * T_);
  float acc = cb[d];
#pragma unroll
  for (int k = 0; k < DC_; ++k) {
    int ts = t - (DC_ - 1) + k;
    if (ts >= 0)
      acc += cw[d * DC_ + k] * xz[((size_t)(b * T_ + ts)) * (2 * DI_) + d];
  }
  xc[(size_t)g] = siluf(acc);
}

// selective scan: one thread per (b, d); state h[DS] in registers, T=32 steps
__global__ void k_scan(const float* __restrict__ xz, const float* __restrict__ xc,
                       const float* __restrict__ dbl, const float* __restrict__ dtb,
                       const float* __restrict__ Alog, const float* __restrict__ Dp,
                       float* __restrict__ y) {
  int g = blockIdx.x * blockDim.x + threadIdx.x;
  if (g >= B_ * DI_) return;
  int d = g % DI_;
  int b = g / DI_;
  float h[DS_], A[DS_];
#pragma unroll
  for (int s = 0; s < DS_; ++s) { h[s] = 0.f; A[s] = -__expf(Alog[d * DS_ + s]); }
  for (int t = 0; t < T_; ++t) {
    size_t base = (size_t)(b * T_ + t);
    float dt = dtb[base * DI_ + d];
    float x  = xc [base * DI_ + d];
    float z  = xz [base * (2 * DI_) + DI_ + d];
    const float* Bm = dbl + base * (DR_ + 2 * DS_) + DR_;
    const float* Cm = Bm + DS_;
    float acc = 0.f;
#pragma unroll
    for (int s = 0; s < DS_; ++s) {
      h[s] = __expf(A[s] * dt) * h[s] + dt * Bm[s] * x;
      acc += h[s] * Cm[s];
    }
    y[base * DI_ + d] = (acc + Dp[d] * x) * siluf(z);
  }
}

// ---------------------------------------------------------------------------
// Generic WMMA bf16 GEMM:  C[M,N] = act(A[M,K(lda)] * W[N,K]^T + bias[N])
// 256 threads = 8 waves; block tile 128x64; BK=32; wave tile 32x32 (2x2 WMMA).
// W tile (64 rows x 32 cols bf16) is staged into LDS by the Tensor Data Mover
// with hardware padding (1 DWORD every 16 DWORDs) matching Bs[64][34].
// ACT: 0=none, 1=relu, 2=softplus
// ---------------------------------------------------------------------------
template <int ACT>
__global__ __launch_bounds__(256) void k_gemm(
    const float* __restrict__ A, int lda,
    const bf16_t* __restrict__ W,
    const float* __restrict__ bias,
    float* __restrict__ C,
    int M, int N, int K) {
  __shared__ bf16_t As[128][34];
  __shared__ bf16_t Bs[64][34];
  const int tid  = threadIdx.x;
  const int lane = tid & 31;
  const int w    = tid >> 5;
  const int wm   = w & 3;   // 4 wave-rows
  const int wn   = w >> 2;  // 2 wave-cols
  const int row0 = blockIdx.x * 128;
  const int col0 = blockIdx.y * 64;

  v8f acc[2][2];
#pragma unroll
  for (int tm = 0; tm < 2; ++tm)
#pragma unroll
    for (int tn = 0; tn < 2; ++tn)
#pragma unroll
      for (int i = 0; i < 8; ++i) acc[tm][tn][i] = 0.f;

  for (int k0 = 0; k0 < K; k0 += 32) {
#if USE_TDM
    // ---- TDM: DMA the 64x32 bf16 W tile into Bs with HW padding ----
    if (w == 0) {
      unsigned long long ga =
          (unsigned long long)(uintptr_t)(W + (size_t)col0 * K + k0);
      unsigned ldso = (unsigned)(uintptr_t)&Bs[0][0]; // generic addr low 32b == LDS offset
      unsigned td0 = (unsigned)(K - k0);              // remaining K (OOB -> zeros)
      unsigned td1 = (unsigned)(N - col0);            // remaining N rows
      v4u g0;
      g0[0] = 1u;                                     // count=1, 2D descriptor
      g0[1] = ldso;                                   // lds_addr
      g0[2] = (unsigned)ga;                           // global_addr[31:0]
      g0[3] = (unsigned)((ga >> 32) & 0x01ffffffu) | 0x80000000u; // addr[56:32] | type=2
      v8i g1;
      // data_size=1(2B) | pad_enable | pad_interval=3(16 DW) | pad_amount=0(1 DW)
      g1[0] = (1 << 16) | (1 << 20) | (3 << 22);
      g1[1] = (int)((td0 & 0xffffu) << 16);           // tensor_dim0[15:0]
      g1[2] = (int)((td0 >> 16) | ((td1 & 0xffffu) << 16)); // td0 hi | td1 lo
      g1[3] = (int)((td1 >> 16) | (32u << 16));       // td1 hi | tile_dim0=32
      g1[4] = 64;                                     // tile_dim1=64, tile_dim2=0
      g1[5] = (int)(unsigned)K;                       // tensor_dim0_stride = K elems
      g1[6] = 0;
      g1[7] = 0;
      v4i z4; z4[0] = 0; z4[1] = 0; z4[2] = 0; z4[3] = 0;
      v8i z8; z8[0] = 0; z8[1] = 0; z8[2] = 0; z8[3] = 0;
      z8[4] = 0; z8[5] = 0; z8[6] = 0; z8[7] = 0;
      __builtin_amdgcn_tensor_load_to_lds(g0, g1, z4, z4, z8, 0);
    }
#else
#pragma unroll
    for (int i = 0; i < 8; ++i) {
      int idx = i * 256 + tid;
      int n = idx >> 5, c = idx & 31;
      bf16_t v = (bf16_t)0.f;
      if (col0 + n < N && k0 + c < K) v = W[(size_t)(col0 + n) * K + k0 + c];
      Bs[n][c] = v;
    }
#endif
    // prefetch next K-slab of A into L2 (global_prefetch_b8)
    if (k0 + 32 < K) {
      int pr = row0 + (tid >> 1);
      if (pr < M) __builtin_prefetch(&A[(size_t)pr * lda + k0 + 32], 0, 0);
    }
    // stage A tile 128x32 (f32 -> bf16) on the VALU path (overlaps TDM)
#pragma unroll
    for (int i = 0; i < 16; ++i) {
      int idx = i * 256 + tid;
      int r = idx >> 5, c = idx & 31;
      float v = 0.f;
      if (row0 + r < M && k0 + c < K) v = A[(size_t)(row0 + r) * lda + k0 + c];
      As[r][c] = (bf16_t)v;
    }
#if USE_TDM
    if (w == 0) __builtin_amdgcn_s_wait_tensorcnt(0);
#endif
    __syncthreads();

    v16bf af[2], bfv[2];
#pragma unroll
    for (int t = 0; t < 2; ++t) {
      int r = wm * 32 + t * 16 + (lane & 15);
      int n = wn * 32 + t * 16 + (lane & 15);
#pragma unroll
      for (int x = 0; x < 16; ++x) {
        int ka = ((lane >> 4) << 4) + x;                          // A frag K map
        af[t][x] = As[r][ka];
        int kb = ((lane >> 4) << 4) + x;                          // B frag K map
        bfv[t][x] = Bs[n][kb];
      }
    }
#pragma unroll
    for (int tm = 0; tm < 2; ++tm)
#pragma unroll
      for (int tn = 0; tn < 2; ++tn)
        acc[tm][tn] = __builtin_amdgcn_wmma_f32_16x16x32_bf16(
            false, af[tm], false, bfv[tn], (short)0, acc[tm][tn], false, false);
    __syncthreads();
  }

  // epilogue: D layout -> global, fuse bias + activation
#pragma unroll
  for (int tm = 0; tm < 2; ++tm)
#pragma unroll
    for (int tn = 0; tn < 2; ++tn) {
      int n = col0 + wn * 32 + tn * 16 + (lane & 15);
      if (n >= N) continue;
      float bv = bias ? bias[n] : 0.f;
#pragma unroll
      for (int i = 0; i < 8; ++i) {
        int m = row0 + wm * 32 + tm * 16 + ((lane >> 4) << 3) + i;
        if (m >= M) continue;
        float v = acc[tm][tn][i] + bv;
        if (ACT == 1) v = v > 0.f ? v : 0.f;
        else if (ACT == 2) v = (v > 20.f) ? v : log1pf(__expf(v));
        C[(size_t)m * N + n] = v;
      }
    }
}

// ---------------------------------------------------------------------------
// Fused attention per (batch, head): scores WMMA -> softmax -> ctx WMMA
// ---------------------------------------------------------------------------
__global__ __launch_bounds__(256) void k_attn(const float* __restrict__ Q,
    const float* __restrict__ Kp, const float* __restrict__ Vp,
    float* __restrict__ ctx) {
  __shared__ bf16_t Qs[T_][HD_ + 8];
  __shared__ bf16_t Cs[64][HD_ + 8];
  __shared__ float  Ss[T_][L_ + 8];
  __shared__ bf16_t Ps[T_][L_ + 8];

  const int bh = blockIdx.x;
  const int b = bh / H_, hh = bh % H_;
  const int tid = threadIdx.x, lane = tid & 31, w = tid >> 5;
  const int tm = w & 1, tn = w >> 1;   // 2 (M) x 4 (N) wave tiles

  // stage Q, folding 1/sqrt(hd)=0.125 into it
  for (int idx = tid; idx < T_ * HD_; idx += 256) {
    int r = idx >> 6, c = idx & 63;
    Qs[r][c] = (bf16_t)(Q[((size_t)(b * T_ + r)) * E_ + hh * HD_ + c] * 0.125f);
  }
  __syncthreads();

  // scores S = Q K^T, 64-key chunks
  for (int c0 = 0; c0 < L_; c0 += 64) {
    for (int idx = tid; idx < 64 * HD_; idx += 256) {
      int r = idx >> 6, c = idx & 63;
      Cs[r][c] = (bf16_t)Kp[((size_t)(b * L_ + c0 + r)) * E_ + hh * HD_ + c];
    }
    __syncthreads();
    v8f acc;
#pragma unroll
    for (int i = 0; i < 8; ++i) acc[i] = 0.f;
#pragma unroll
    for (int ks = 0; ks < HD_; ks += 32) {
      v16bf a, bfv;
      int ar = tm * 16 + (lane & 15);
      int bn = tn * 16 + (lane & 15);
#pragma unroll
      for (int x = 0; x < 16; ++x) {
        int ka = ks + ((lane >> 4) << 4) + x;
        a[x] = Qs[ar][ka];
        int kb = ks + ((lane >> 4) << 4) + x;
        bfv[x] = Cs[bn][kb];
      }
      acc = __builtin_amdgcn_wmma_f32_16x16x32_bf16(false, a, false, bfv,
                                                    (short)0, acc, false, false);
    }
#pragma unroll
    for (int i = 0; i < 8; ++i) {
      int m = tm * 16 + ((lane >> 4) << 3) + i;
      int n = c0 + tn * 16 + (lane & 15);
      Ss[m][n] = acc[i];
    }
    __syncthreads();
  }

  // row softmax -> P (bf16)
  if (tid < T_) {
    int r = tid;
    float mx = -1e30f;
    for (int c = 0; c < L_; ++c) mx = fmaxf(mx, Ss[r][c]);
    float sum = 0.f;
    for (int c = 0; c < L_; ++c) { float e = __expf(Ss[r][c] - mx); Ss[r][c] = e; sum += e; }
    float inv = 1.f / sum;
    for (int c = 0; c < L_; ++c) Ps[r][c] = (bf16_t)(Ss[r][c] * inv);
  }
  __syncthreads();

  // ctx = P V, accumulate across 64-key chunks
  v8f acc;
#pragma unroll
  for (int i = 0; i < 8; ++i) acc[i] = 0.f;
  for (int c0 = 0; c0 < L_; c0 += 64) {
    for (int idx = tid; idx < 64 * HD_; idx += 256) {
      int r = idx >> 6, c = idx & 63;
      Cs[r][c] = (bf16_t)Vp[((size_t)(b * L_ + c0 + r)) * E_ + hh * HD_ + c];
    }
    __syncthreads();
#pragma unroll
    for (int ks = 0; ks < 64; ks += 32) {
      v16bf a, bfv;
      int ar = tm * 16 + (lane & 15);
      int fn = tn * 16 + (lane & 15);
#pragma unroll
      for (int x = 0; x < 16; ++x) {
        int ka = c0 + ks + ((lane >> 4) << 4) + x;
        a[x] = Ps[ar][ka];
        int kb = ks + ((lane >> 4) << 4) + x;
        bfv[x] = Cs[kb][fn];
      }
      acc = __builtin_amdgcn_wmma_f32_16x16x32_bf16(false, a, false, bfv,
                                                    (short)0, acc, false, false);
    }
    __syncthreads();
  }
#pragma unroll
  for (int i = 0; i < 8; ++i) {
    int m = tm * 16 + ((lane >> 4) << 3) + i;
    int f = tn * 16 + (lane & 15);
    ctx[((size_t)(b * T_ + m)) * E_ + hh * HD_ + f] = acc[i];
  }
}

// ---------------------------------------------------------------------------
extern "C" void kernel_launch(void* const* d_in, const int* in_sizes, int n_in,
                              void* d_out, int out_size, void* d_ws, size_t ws_size,
                              hipStream_t stream) {
  (void)in_sizes; (void)n_in; (void)out_size; (void)ws_size;

  const float* enc      = (const float*)d_in[0];
  const float* qe       = (const float*)d_in[1];
  const float* pe       = (const float*)d_in[2];
  const float* m_norm_w = (const float*)d_in[3];
  const float* m_in_w   = (const float*)d_in[4];
  const float* m_conv_w = (const float*)d_in[5];
  const float* m_conv_b = (const float*)d_in[6];
  const float* m_xproj  = (const float*)d_in[7];
  const float* m_dt_w   = (const float*)d_in[8];
  const float* m_dt_b   = (const float*)d_in[9];
  const float* m_Alog   = (const float*)d_in[10];
  const float* m_D      = (const float*)d_in[11];
  const float* m_out_w  = (const float*)d_in[12];
  const float* rms_w    = (const float*)d_in[13];
  const float* wq = (const float*)d_in[14]; const float* bq = (const float*)d_in[15];
  const float* wk = (const float*)d_in[16]; const float* bk = (const float*)d_in[17];
  const float* wv = (const float*)d_in[18]; const float* bv = (const float*)d_in[19];
  const float* wo = (const float*)d_in[20]; const float* bo = (const float*)d_in[21];
  const float* ln1_w = (const float*)d_in[22]; const float* ln1_b = (const float*)d_in[23];
  const float* f_w1 = (const float*)d_in[24]; const float* f_b1 = (const float*)d_in[25];
  const float* f_w2 = (const float*)d_in[26]; const float* f_b2 = (const float*)d_in[27];
  const float* ln2_w = (const float*)d_in[28]; const float* ln2_b = (const float*)d_in[29];
  const float* out_w = (const float*)d_in[30]; const float* out_b = (const float*)d_in[31];

  char* ws = (char*)d_ws;
  size_t off = 0;
  auto alloc = [&](size_t bytes) -> void* {
    void* p = ws + off;
    off += (bytes + 255) & ~(size_t)255;
    return p;
  };

  const int M  = B_ * T_;   // 2048
  const int MK = B_ * L_;   // 16384

  // bf16 weight mirrors
  bf16_t* wb_in = (bf16_t*)alloc((size_t)ND_ * 2 * DI_ * E_ * 2);
  bf16_t* wb_xp = (bf16_t*)alloc((size_t)ND_ * (DR_ + 2 * DS_) * DI_ * 2);
  bf16_t* wb_dt = (bf16_t*)alloc((size_t)ND_ * DI_ * DR_ * 2);
  bf16_t* wb_om = (bf16_t*)alloc((size_t)ND_ * E_ * DI_ * 2);
  bf16_t* wb_q  = (bf16_t*)alloc((size_t)E_ * E_ * 2);
  bf16_t* wb_k  = (bf16_t*)alloc((size_t)E_ * E_ * 2);
  bf16_t* wb_v  = (bf16_t*)alloc((size_t)E_ * E_ * 2);
  bf16_t* wb_o  = (bf16_t*)alloc((size_t)E_ * E_ * 2);
  bf16_t* wb_f1 = (bf16_t*)alloc((size_t)HID_ * E_ * 2);
  bf16_t* wb_f2 = (bf16_t*)alloc((size_t)E_ * HID_ * 2);
  bf16_t* wb_lg = (bf16_t*)alloc((size_t)V_ * E_ * 2);

  // f32 activations
  float* hidden   = (float*)alloc((size_t)M * E_ * 4);
  float* residual = (float*)alloc((size_t)M * E_ * 4);
  float* normed   = (float*)alloc((size_t)M * E_ * 4);
  float* xz       = (float*)alloc((size_t)M * 2 * DI_ * 4);
  float* xc       = (float*)alloc((size_t)M * DI_ * 4);
  float* dblb     = (float*)alloc((size_t)M * (DR_ + 2 * DS_) * 4);
  float* dtb      = (float*)alloc((size_t)M * DI_ * 4);
  float* ybuf     = (float*)alloc((size_t)M * DI_ * 4);
  float* qproj    = (float*)alloc((size_t)M * E_ * 4);
  float* kproj    = (float*)alloc((size_t)MK * E_ * 4);
  float* vproj    = (float*)alloc((size_t)MK * E_ * 4);
  float* ctxb     = (float*)alloc((size_t)M * E_ * 4);
  float* tmp      = (float*)alloc((size_t)M * E_ * 4);
  float* ff       = (float*)alloc((size_t)M * HID_ * 4);

  auto cvt = [&](const float* s, bf16_t* d, int n) {
    k_cvt<<<(n + 255) / 256, 256, 0, stream>>>(s, d, n);
  };
  auto gemm = [&](int act, const float* A, int lda, const bf16_t* W,
                  const float* bias, float* C, int m, int n, int k) {
    dim3 g((m + 127) / 128, (n + 63) / 64);
    if (act == 0)      k_gemm<0><<<g, 256, 0, stream>>>(A, lda, W, bias, C, m, n, k);
    else if (act == 1) k_gemm<1><<<g, 256, 0, stream>>>(A, lda, W, bias, C, m, n, k);
    else               k_gemm<2><<<g, 256, 0, stream>>>(A, lda, W, bias, C, m, n, k);
  };

  // --- weight conversion (once per launch; deterministic) ---
  cvt(m_in_w,  wb_in, ND_ * 2 * DI_ * E_);
  cvt(m_xproj, wb_xp, ND_ * (DR_ + 2 * DS_) * DI_);
  cvt(m_dt_w,  wb_dt, ND_ * DI_ * DR_);
  cvt(m_out_w, wb_om, ND_ * E_ * DI_);
  cvt(wq, wb_q, E_ * E_);  cvt(wk, wb_k, E_ * E_);
  cvt(wv, wb_v, E_ * E_);  cvt(wo, wb_o, E_ * E_);
  cvt(f_w1, wb_f1, HID_ * E_);  cvt(f_w2, wb_f2, E_ * HID_);
  cvt(out_w, wb_lg, V_ * E_);

  // --- init query stream ---
  k_init<<<(M * E_ + 255) / 256, 256, 0, stream>>>(qe, pe, hidden, residual);

  // --- 12 Mamba layers ---
  for (int l = 0; l < ND_; ++l) {
    k_add_rmsnorm<<<M, 256, 0, stream>>>(residual, hidden, m_norm_w + (size_t)l * E_, normed);
    gemm(0, normed, E_, wb_in + (size_t)l * 2 * DI_ * E_, nullptr, xz, M, 2 * DI_, E_);
    k_conv_silu<<<(M * DI_ + 255) / 256, 256, 0, stream>>>(
        xz, m_conv_w + (size_t)l * DI_ * DC_, m_conv_b + (size_t)l * DI_, xc);
    gemm(0, xc, DI_, wb_xp + (size_t)l * (DR_ + 2 * DS_) * DI_, nullptr,
         dblb, M, DR_ + 2 * DS_, DI_);
    gemm(2, dblb, DR_ + 2 * DS_, wb_dt + (size_t)l * DI_ * DR_,
         m_dt_b + (size_t)l * DI_, dtb, M, DI_, DR_);
    k_scan<<<(B_ * DI_ + 255) / 256, 256, 0, stream>>>(
        xz, xc, dblb, dtb, m_Alog + (size_t)l * DI_ * DS_, m_D + (size_t)l * DI_, ybuf);
    gemm(0, ybuf, DI_, wb_om + (size_t)l * E_ * DI_, nullptr, hidden, M, E_, DI_);
  }

  // --- final mamba-stack norm ---
  k_add_rmsnorm<<<M, 256, 0, stream>>>(residual, hidden, rms_w, normed);

  // --- cross attention ---
  gemm(0, normed, E_, wb_q, bq, qproj, M, E_, E_);
  gemm(0, enc,    E_, wb_k, bk, kproj, MK, E_, E_);
  gemm(0, enc,    E_, wb_v, bv, vproj, MK, E_, E_);
  k_attn<<<B_ * H_, 256, 0, stream>>>(qproj, kproj, vproj, ctxb);
  gemm(0, ctxb, E_, wb_o, bo, tmp, M, E_, E_);
  k_add<<<(M * E_ + 255) / 256, 256, 0, stream>>>(residual, tmp, M * E_);
  k_layernorm<<<M, 256, 0, stream>>>(residual, ln1_w, ln1_b, normed);

  // --- FFN ---
  gemm(1, normed, E_, wb_f1, f_b1, ff, M, HID_, E_);
  gemm(0, ff, HID_, wb_f2, f_b2, tmp, M, E_, HID_);
  k_add<<<(M * E_ + 255) / 256, 256, 0, stream>>>(residual, tmp, M * E_);
  k_layernorm<<<M, 256, 0, stream>>>(residual, ln2_w, ln2_b, normed);

  // --- logits ---
  gemm(0, normed, E_, wb_lg, out_b, (float*)d_out, M, V_, E_);
}
